// NMSquaredGaussianMixture_69509750718885
// MI455X (gfx1250) — matrix operand order repaired and checked by reading, not compiled, not run on hardware
//
#include <hip/hip_runtime.h>
#include <math.h>

typedef __attribute__((ext_vector_type(16))) _Float16 v16h;
typedef __attribute__((ext_vector_type(8)))  float    v8f;

#define N_PTS 131072
#define KCL   32
#define DIM   32
#define LOG2PI 1.8378770664093453f

// ws layout (bytes):
//   0    double total  (sum of s_n^2)
//   8    double zacc   (pairwise normalizer z)
//   64   float  coef[32]          = -0.5*(logdet_k + D*log(2pi))
//   256  float  negb[32*32]       = -(L_k^{-1} mu_k)
//   8192 _Float16 AinvH[32*2*32*16]  A-fragments of L_k^{-1}, WMMA layout (64KB)
// total: 73728 bytes

// ---------------- prep 1: per-cluster Cholesky, inverse, fragments ----------
__global__ void prep1_kernel(const float* __restrict__ means,
                             const float* __restrict__ chols,
                             void* ws)
{
    int k = threadIdx.x;
    double* acc = (double*)ws;
    if (k == 0) { acc[0] = 0.0; acc[1] = 0.0; }   // re-zero every launch
    if (k >= KCL) return;

    float*     coef  = (float*)((char*)ws + 64);
    float*     negb  = (float*)((char*)ws + 256);
    _Float16*  AinvH = (_Float16*)((char*)ws + 8192);

    // sigma_k = C C^T + I
    float S[DIM*DIM];
    const float* C = chols + k*DIM*DIM;
    for (int i = 0; i < DIM; ++i)
        for (int j = 0; j <= i; ++j) {
            float s = (i == j) ? 1.0f : 0.0f;
            for (int l = 0; l < DIM; ++l) s += C[i*DIM+l]*C[j*DIM+l];
            S[i*DIM+j] = s; S[j*DIM+i] = s;
        }
    // in-place lower Cholesky
    float logdet = 0.f;
    for (int j = 0; j < DIM; ++j) {
        float d = S[j*DIM+j];
        for (int p = 0; p < j; ++p) d -= S[j*DIM+p]*S[j*DIM+p];
        d = sqrtf(d);
        S[j*DIM+j] = d;
        logdet += 2.0f*logf(d);
        float inv = 1.0f/d;
        for (int i = j+1; i < DIM; ++i) {
            float v = S[i*DIM+j];
            for (int p = 0; p < j; ++p) v -= S[i*DIM+p]*S[j*DIM+p];
            S[i*DIM+j] = v*inv;
        }
    }
    coef[k] = -0.5f*(logdet + (float)DIM*LOG2PI);

    // Ai = L^{-1} (lower triangular)
    float Ai[DIM*DIM];
    for (int c = 0; c < DIM; ++c)
        for (int r = 0; r < DIM; ++r) {
            if (r < c) { Ai[r*DIM+c] = 0.f; continue; }
            float v = (r == c) ? 1.f : 0.f;
            for (int p = c; p < r; ++p) v -= S[r*DIM+p]*Ai[p*DIM+c];
            Ai[r*DIM+c] = v / S[r*DIM+r];
        }

    // negb = -(Ai * mu)
    const float* mu = means + k*DIM;
    for (int r = 0; r < DIM; ++r) {
        float v = 0.f;
        for (int c = 0; c <= r; ++c) v += Ai[r*DIM+c]*mu[c];
        negb[k*DIM + r] = -v;
    }

    // store Ai as two 16x32 f16 A-fragments with contiguous k per lane half (same k order as B):
// lane<16: M=lane, j=0..15 -> K=0..15
// lane>=16: M=lane-16, j=0..15 -> K=16..31
    for (int f = 0; f < 2; ++f)
        for (int lane = 0; lane < 32; ++lane) {
            int m    = (lane & 15) + f*16;
            int kofs = (lane >= 16) ? 16 : 0;
            for (int j = 0; j < 16; ++j) {
                int kd = j + kofs;
                AinvH[(((k*2 + f)*32) + lane)*16 + j] = (_Float16)Ai[m*DIM + kd];
            }
        }
}

// ---------------- prep 2: pairwise normalizer z -----------------------------
__global__ void prep2_kernel(const float* __restrict__ means,
                             const float* __restrict__ chols,
                             const float* __restrict__ weights,
                             void* ws)
{
    int t = blockIdx.x*blockDim.x + threadIdx.x;
    if (t >= KCL*KCL) return;
    int i = t / KCL, j = t % KCL;
    const float* Ci = chols + i*DIM*DIM;
    const float* Cj = chols + j*DIM*DIM;

    // S = sigma_i + sigma_j  (lower triangle), diag gets +2 from the two I's
    float S[DIM*DIM];
    for (int r = 0; r < DIM; ++r)
        for (int c = 0; c <= r; ++c) {
            float s = (r == c) ? 2.0f : 0.0f;
            for (int l = 0; l < DIM; ++l)
                s += Ci[r*DIM+l]*Ci[c*DIM+l] + Cj[r*DIM+l]*Cj[c*DIM+l];
            S[r*DIM+c] = s;
        }
    // Cholesky (lower only)
    float logdet = 0.f;
    for (int cc = 0; cc < DIM; ++cc) {
        float d = S[cc*DIM+cc];
        for (int p = 0; p < cc; ++p) d -= S[cc*DIM+p]*S[cc*DIM+p];
        d = sqrtf(d);
        S[cc*DIM+cc] = d;
        logdet += 2.0f*logf(d);
        float inv = 1.0f/d;
        for (int r = cc+1; r < DIM; ++r) {
            float v = S[r*DIM+cc];
            for (int p = 0; p < cc; ++p) v -= S[r*DIM+p]*S[cc*DIM+p];
            S[r*DIM+cc] = v*inv;
        }
    }
    // forward solve L y = (mu_i - mu_j)
    float y[DIM];
    float maha = 0.f;
    for (int r = 0; r < DIM; ++r) {
        float v = means[i*DIM+r] - means[j*DIM+r];
        for (int p = 0; p < r; ++p) v -= S[r*DIM+p]*y[p];
        y[r] = v / S[r*DIM+r];
        maha += y[r]*y[r];
    }
    float Zij = __expf(-0.5f*(maha + logdet + (float)DIM*LOG2PI));
    double contrib = (double)(weights[i]*weights[j]*Zij);
    atomicAdd(((double*)ws) + 1, contrib);
}

// ---------------- main: WMMA batched maha + density accumulation ------------
#define NT 4   // 4 x 16-point tiles per wave = 64 points/wave

__global__ __launch_bounds__(256) void main_kernel(const float* __restrict__ X,
                                                   const float* __restrict__ weights,
                                                   void* ws)
{
    const float* coef = (const float*)((char*)ws + 64);
    const float* negb = (const float*)((char*)ws + 256);
    const v16h*  AinvH = (const v16h*)((char*)ws + 8192);
    double* acc = (double*)ws;

    int lane = threadIdx.x & 31;
    int wave = threadIdx.x >> 5;
    int base = (blockIdx.x*8 + wave) * (NT*16);
    bool hi  = lane >= 16;
    int pcol = lane & 15;
    int dinBase = hi ? 16 : 0;

    // B fragments: X^T tiles (cluster independent; built once, reused 32x).
    // B layout (32x16 f16): lane<16 -> col n=lane, K=0..15 packed ascending;
    //                       lane>=16 -> col n=lane-16, K=16..31.
    v16h B[NT];
    #pragma unroll
    for (int t = 0; t < NT; ++t) {
        const float* xr = X + (size_t)(base + t*16 + pcol)*DIM + dinBase;
        #pragma unroll
        for (int jj = 0; jj < 16; ++jj) B[t][jj] = (_Float16)xr[jj];
    }

    float sacc[NT];
    #pragma unroll
    for (int t = 0; t < NT; ++t) sacc[t] = 0.f;

    for (int k = 0; k < KCL; ++k) {
        // A fragments: rows 0..15 and 16..31 of L_k^{-1}
        v16h A0 = AinvH[(k*2 + 0)*32 + lane];
        v16h A1 = AinvH[(k*2 + 1)*32 + lane];
        // C = broadcast of -b_k along columns: D = Ainv*X^T - b  (= y directly)
        const float* nb = negb + k*DIM;
        v8f C0, C1;
        #pragma unroll
        for (int r = 0; r < 8; ++r) {
            C0[r] = hi ? nb[r + 8]      : nb[r];
            C1[r] = hi ? nb[16 + r + 8] : nb[16 + r];
        }
        float wk = weights[k];
        float ck = coef[k];
        #pragma unroll
        for (int t = 0; t < NT; ++t) {
            v8f d0 = __builtin_amdgcn_wmma_f32_16x16x32_f16(
                false, A0, false, B[t], (short)0, C0, false, false);
            v8f d1 = __builtin_amdgcn_wmma_f32_16x16x32_f16(
                false, A1, false, B[t], (short)0, C1, false, false);
            // lane holds one point; sum y^2 over the 16 output dims it owns
            float m = 0.f;
            #pragma unroll
            for (int r = 0; r < 8; ++r) m += d0[r]*d0[r] + d1[r]*d1[r];
            m += __shfl_xor(m, 16, 32);            // combine the two lane halves
            sacc[t] += wk * __expf(ck - 0.5f*m);   // s_n accumulation
        }
    }

    // sum of s^2 over this wave's points (halves hold duplicates -> *0.5)
    double q = 0.0;
    #pragma unroll
    for (int t = 0; t < NT; ++t) q += (double)sacc[t]*(double)sacc[t];
    #pragma unroll
    for (int m = 16; m >= 1; m >>= 1) q += __shfl_xor(q, m, 32);
    q *= 0.5;
    if (lane == 0) atomicAdd(acc, q);
}

// ---------------- finalize --------------------------------------------------
__global__ void finalize_kernel(void* ws, float* out)
{
    double total = ((double*)ws)[0];
    double z     = ((double*)ws)[1];
    out[0] = (float)(-(log(total) - log(z)) / (double)N_PTS);
}

extern "C" void kernel_launch(void* const* d_in, const int* in_sizes, int n_in,
                              void* d_out, int out_size, void* d_ws, size_t ws_size,
                              hipStream_t stream) {
    const float* X       = (const float*)d_in[0];
    const float* means   = (const float*)d_in[1];
    const float* chols   = (const float*)d_in[2];
    const float* weights = (const float*)d_in[3];
    float* out = (float*)d_out;

    prep1_kernel<<<1, 32, 0, stream>>>(means, chols, d_ws);
    prep2_kernel<<<4, 256, 0, stream>>>(means, chols, weights, d_ws);
    // 256 blocks * 8 waves * 64 points = 131072
    main_kernel<<<256, 256, 0, stream>>>(X, weights, d_ws);
    finalize_kernel<<<1, 1, 0, stream>>>(d_ws, out);
}
